// Graphormer_85358180041573
// MI455X (gfx1250) — compile-verified, matrix operand order checked by reading, not run on hardware
//
#include <hip/hip_runtime.h>
#include <hip/hip_bf16.h>

// TransformerConv-style GNN for MI455X (gfx1250, wave32, WMMA).
// GEMMs on v_wmma_f32_16x16x32_f16 (f32 accumulate, bias folded into C).
// Weights pre-swizzled into B-lane-format in LDS so each B operand is a
// single 32B vector LDS load instead of 16 scalar ds_load_u16.
// Graph stages (segment softmax / scatter) use f32 global atomics; all node
// tensors (12.8 MB each) are L2-resident on the 192 MB L2.

#define N_NODES 50000
#define N_EDGES 800000
#define HID     64
#define HEADS   4
#define DH      16
#define NCLS    4

typedef __attribute__((ext_vector_type(16))) _Float16 v16h;
typedef __attribute__((ext_vector_type(8)))  _Float16 v8h;
typedef __attribute__((ext_vector_type(8)))  float    v8f;

// CDNA5 16-bit A-matrix 16x32 lane layout: lane half 'hi' holds K = hi*8+j
// (j<8) in VGPR0..3 and K = 16+hi*8+(j-8) in VGPR4..7.
__device__ __forceinline__ int kmap(int hi, int j) {
    return (j < 8) ? (hi * 8 + j) : (16 + hi * 8 + (j - 8));
}

// order-preserving float <-> u32 encoding for atomic max
__device__ __forceinline__ unsigned encf(float f) {
    unsigned u = __float_as_uint(f);
    return (u & 0x80000000u) ? ~u : (u | 0x80000000u);
}
__device__ __forceinline__ float decf(unsigned u) {
    unsigned b = (u & 0x80000000u) ? (u & 0x7FFFFFFFu) : ~u;
    return __uint_as_float(b);
}

// Load one K-chunk (32 wide) of a row-major f32 row as an f16 A-operand:
// lane's K values are two contiguous runs of 8 floats -> 4x float4 loads.
__device__ __forceinline__ v16h load_a_chunk_f32(const float* row, int hi, int c) {
    const float4* p0 = (const float4*)(row + c * 32 + hi * 8);
    const float4* p1 = (const float4*)(row + 16 + c * 32 + hi * 8);
    float4 f0 = p0[0], f1 = p0[1], f2 = p1[0], f3 = p1[1];
    v16h a;
    a[0]  = (_Float16)f0.x; a[1]  = (_Float16)f0.y; a[2]  = (_Float16)f0.z; a[3]  = (_Float16)f0.w;
    a[4]  = (_Float16)f1.x; a[5]  = (_Float16)f1.y; a[6]  = (_Float16)f1.z; a[7]  = (_Float16)f1.w;
    a[8]  = (_Float16)f2.x; a[9]  = (_Float16)f2.y; a[10] = (_Float16)f2.z; a[11] = (_Float16)f2.w;
    a[12] = (_Float16)f3.x; a[13] = (_Float16)f3.y; a[14] = (_Float16)f3.z; a[15] = (_Float16)f3.w;
    return a;
}

// Same for summing two gathered rows (final kernel: out[src]+out[dst]).
__device__ __forceinline__ v16h load_a_chunk_sum(const float* ra, const float* rb, int hi, int c) {
    const float4* a0 = (const float4*)(ra + c * 32 + hi * 8);
    const float4* a1 = (const float4*)(ra + 16 + c * 32 + hi * 8);
    const float4* b0 = (const float4*)(rb + c * 32 + hi * 8);
    const float4* b1 = (const float4*)(rb + 16 + c * 32 + hi * 8);
    float4 f0 = a0[0], f1 = a0[1], f2 = a1[0], f3 = a1[1];
    float4 g0 = b0[0], g1 = b0[1], g2 = b1[0], g3 = b1[1];
    v16h a;
    a[0]  = (_Float16)(f0.x + g0.x); a[1]  = (_Float16)(f0.y + g0.y);
    a[2]  = (_Float16)(f0.z + g0.z); a[3]  = (_Float16)(f0.w + g0.w);
    a[4]  = (_Float16)(f1.x + g1.x); a[5]  = (_Float16)(f1.y + g1.y);
    a[6]  = (_Float16)(f1.z + g1.z); a[7]  = (_Float16)(f1.w + g1.w);
    a[8]  = (_Float16)(f2.x + g2.x); a[9]  = (_Float16)(f2.y + g2.y);
    a[10] = (_Float16)(f2.z + g2.z); a[11] = (_Float16)(f2.w + g2.w);
    a[12] = (_Float16)(f3.x + g3.x); a[13] = (_Float16)(f3.y + g3.y);
    a[14] = (_Float16)(f3.z + g3.z); a[15] = (_Float16)(f3.w + g3.w);
    return a;
}

// ---------------------------------------------------------------------------
// Kernel 1: h = x@Wn+bn ; q/k/v = h@W*+b* ; oacc = h@Ws+bs (skip term).
// One wave per 16-row node tile, 8 waves (256 thr) per block.
// Weights staged in LDS pre-swizzled into B-lane-format:
//   index = mtx*4096 + nt*1024 + chunk*512 + lane*16 + j
//   maps to W[(chunk*32 + kmap(hi,j))*64 + nt*16 + lr]
// so each B operand is one contiguous 32B LDS vector load.
// ---------------------------------------------------------------------------
__global__ __launch_bounds__(256) void node_gemm_kernel(
    const float* __restrict__ x,
    const float* __restrict__ Wn, const float* __restrict__ bn,
    const float* __restrict__ Wq, const float* __restrict__ bq,
    const float* __restrict__ Wk, const float* __restrict__ bk,
    const float* __restrict__ Wv, const float* __restrict__ bv,
    const float* __restrict__ Ws, const float* __restrict__ bs,
    float* __restrict__ qout, float* __restrict__ kout,
    float* __restrict__ vout, float* __restrict__ oout)
{
    __shared__ __attribute__((aligned(32))) _Float16 wlds[5 * HID * HID]; // 40KB
    __shared__ __attribute__((aligned(32))) _Float16 hlds[8][16 * HID];   // 16KB

    const int tid = threadIdx.x;
    const float* wsrc[5] = {Wn, Wq, Wk, Wv, Ws};
    for (int i = tid; i < 5 * HID * HID; i += 256) {
        int mtx = i >> 12;
        int rem = i & 4095;
        int nt = rem >> 10;
        int ch = (rem >> 9) & 1;
        int ln = (rem >> 4) & 31;
        int j  = i & 15;
        int col = nt * 16 + (ln & 15);
        int k   = ch * 32 + kmap(ln >> 4, j);
        wlds[i] = (_Float16)wsrc[mtx][k * HID + col];
    }
    __syncthreads();

    const int wave = tid >> 5;
    const int lane = tid & 31;
    const int hi   = lane >> 4;
    const int lr   = lane & 15;
    const int tile = blockIdx.x * 8 + wave;
    const bool valid = tile < (N_NODES / 16);
    const int row0 = tile * 16;

    if (valid) {
        const float* xr = x + (size_t)(row0 + lr) * HID;
        __builtin_prefetch(xr + 16 * HID, 0, 1);   // next tile's row
        v16h a0 = load_a_chunk_f32(xr, hi, 0);
        v16h a1 = load_a_chunk_f32(xr, hi, 1);
        // h = x@Wn + bn, 4 N-tiles of 16 columns; stash h tile in LDS
        #pragma unroll
        for (int nt = 0; nt < 4; ++nt) {
            int col = nt * 16 + lr;
            float bias = bn[col];
            v8f c;
            #pragma unroll
            for (int r = 0; r < 8; ++r) c[r] = bias;
            v16h b0 = *(const v16h*)&wlds[nt * 1024 + lane * 16];
            v16h b1 = *(const v16h*)&wlds[nt * 1024 + 512 + lane * 16];
            c = __builtin_amdgcn_wmma_f32_16x16x32_f16(false, a0, false, b0, (short)0, c, false, false);
            c = __builtin_amdgcn_wmma_f32_16x16x32_f16(false, a1, false, b1, (short)0, c, false, false);
            #pragma unroll
            for (int r = 0; r < 8; ++r) {
                int m = r + 8 * hi;                    // C-layout row
                hlds[wave][m * HID + col] = (_Float16)c[r];   // transpose store
            }
        }
    }
    __syncthreads();
    if (!valid) return;

    // re-read h from LDS in A-format: contiguous runs of 8 halfs -> v8h loads
    const _Float16* hr = &hlds[wave][lr * HID];
    v8h l0 = *(const v8h*)(hr + hi * 8);
    v8h l1 = *(const v8h*)(hr + 16 + hi * 8);
    v8h l2 = *(const v8h*)(hr + 32 + hi * 8);
    v8h l3 = *(const v8h*)(hr + 48 + hi * 8);
    v16h ha0 = __builtin_shufflevector(l0, l1, 0,1,2,3,4,5,6,7,8,9,10,11,12,13,14,15);
    v16h ha1 = __builtin_shufflevector(l2, l3, 0,1,2,3,4,5,6,7,8,9,10,11,12,13,14,15);

    const float* bias_ptr[4] = {bq, bk, bv, bs};
    float*       out_ptr[4]  = {qout, kout, vout, oout};
    #pragma unroll
    for (int mtx = 0; mtx < 4; ++mtx) {
        const _Float16* w = &wlds[(mtx + 1) * 4096];
        #pragma unroll
        for (int nt = 0; nt < 4; ++nt) {
            int col = nt * 16 + lr;
            float bias = bias_ptr[mtx][col];
            v8f c;
            #pragma unroll
            for (int r = 0; r < 8; ++r) c[r] = bias;
            v16h b0 = *(const v16h*)&w[nt * 1024 + lane * 16];
            v16h b1 = *(const v16h*)&w[nt * 1024 + 512 + lane * 16];
            c = __builtin_amdgcn_wmma_f32_16x16x32_f16(false, ha0, false, b0, (short)0, c, false, false);
            c = __builtin_amdgcn_wmma_f32_16x16x32_f16(false, ha1, false, b1, (short)0, c, false, false);
            #pragma unroll
            for (int r = 0; r < 8; ++r) {
                int m = r + 8 * hi;
                out_ptr[mtx][(size_t)(row0 + m) * HID + col] = c[r];
            }
        }
    }
}

// ---------------------------------------------------------------------------
// init: zero segment-max (encoded) and segment-sum buffers
// ---------------------------------------------------------------------------
__global__ void init_kernel(unsigned* __restrict__ me, float* __restrict__ ss, int n) {
    int i = blockIdx.x * blockDim.x + threadIdx.x;
    if (i < n) { me[i] = 0u; ss[i] = 0.0f; }
}

// ---------------------------------------------------------------------------
// per-(edge,head) attention logit + segment max via encoded u32 atomicMax
// ---------------------------------------------------------------------------
__global__ void logits_kernel(const int* __restrict__ ei,
                              const float* __restrict__ q, const float* __restrict__ k,
                              float* __restrict__ logits, unsigned* __restrict__ me)
{
    int t = blockIdx.x * blockDim.x + threadIdx.x;
    if (t >= N_EDGES * HEADS) return;
    int e = t >> 2, hd = t & 3;
    int s = ei[e], d = ei[N_EDGES + e];
    const float4* qr = (const float4*)(q + (size_t)d * HID + hd * DH);
    const float4* kr = (const float4*)(k + (size_t)s * HID + hd * DH);
    float acc = 0.f;
    #pragma unroll
    for (int i = 0; i < 4; ++i) {
        float4 qa = qr[i], ka = kr[i];
        acc += qa.x * ka.x + qa.y * ka.y + qa.z * ka.z + qa.w * ka.w;
    }
    acc *= 0.25f;                      // 1/sqrt(D_HEAD)
    logits[t] = acc;
    atomicMax(&me[d * HEADS + hd], encf(acc));
}

// ---------------------------------------------------------------------------
// exp(logit - max) + segment sum
// ---------------------------------------------------------------------------
__global__ void expsum_kernel(const int* __restrict__ ei,
                              float* __restrict__ logits,
                              const unsigned* __restrict__ me,
                              float* __restrict__ ss)
{
    int t = blockIdx.x * blockDim.x + threadIdx.x;
    if (t >= N_EDGES * HEADS) return;
    int e = t >> 2, hd = t & 3;
    int d = ei[N_EDGES + e];
    float m = decf(me[d * HEADS + hd]);
    float ev = __expf(logits[t] - m);
    logits[t] = ev;
    atomicAdd(&ss[d * HEADS + hd], ev);
}

// ---------------------------------------------------------------------------
// message: (v[src] + edge_attr@We + be) * alpha scattered to out[dst]
// (ea computed on the fly: K=2, just 2 FMAs per element)
// ---------------------------------------------------------------------------
__global__ void message_kernel(const int* __restrict__ ei,
                               const float* __restrict__ eattr,
                               const float* __restrict__ We, const float* __restrict__ be,
                               const float* __restrict__ v,
                               const float* __restrict__ logits,
                               const float* __restrict__ ss,
                               float* __restrict__ out)
{
    int t = blockIdx.x * blockDim.x + threadIdx.x;
    if (t >= N_EDGES * HEADS) return;
    int e = t >> 2, hd = t & 3;
    int s = ei[e], d = ei[N_EDGES + e];
    float alpha = logits[t] / (ss[d * HEADS + hd] + 1e-16f);
    float ea0 = eattr[e * 2], ea1 = eattr[e * 2 + 1];
    const float4* w0 = (const float4*)(We + hd * DH);          // We row 0
    const float4* w1 = (const float4*)(We + HID + hd * DH);    // We row 1
    const float4* bb = (const float4*)(be + hd * DH);
    const float4* vr = (const float4*)(v + (size_t)s * HID + hd * DH);
    float* orow = out + (size_t)d * HID + hd * DH;
    #pragma unroll
    for (int i = 0; i < 4; ++i) {
        float4 vv = vr[i], a = w0[i], b = w1[i], c = bb[i];
        float4 m;
        m.x = (vv.x + fmaf(ea0, a.x, fmaf(ea1, b.x, c.x))) * alpha;
        m.y = (vv.y + fmaf(ea0, a.y, fmaf(ea1, b.y, c.y))) * alpha;
        m.z = (vv.z + fmaf(ea0, a.z, fmaf(ea1, b.z, c.z))) * alpha;
        m.w = (vv.w + fmaf(ea0, a.w, fmaf(ea1, b.w, c.w))) * alpha;
        atomicAdd(&orow[i * 4 + 0], m.x);
        atomicAdd(&orow[i * 4 + 1], m.y);
        atomicAdd(&orow[i * 4 + 2], m.z);
        atomicAdd(&orow[i * 4 + 3], m.w);
    }
}

// ---------------------------------------------------------------------------
// final: res = (out[src]+out[dst]) @ Wf + bf via WMMA (Wf zero-padded 4->16,
// staged in LDS in B-lane-format once per block).
// one wave per 16-edge tile; 800000/16 = 50000 tiles exactly
// ---------------------------------------------------------------------------
__global__ __launch_bounds__(256) void final_kernel(
    const int* __restrict__ ei, const float* __restrict__ out,
    const float* __restrict__ Wf, const float* __restrict__ bf,
    float* __restrict__ res)
{
    __shared__ __attribute__((aligned(32))) _Float16 wf_lds[2 * 512]; // 2 chunks

    const int tid = threadIdx.x;
    for (int i = tid; i < 1024; i += 256) {
        int ch = i >> 9;
        int ln = (i >> 4) & 31;
        int j  = i & 15;
        int col = ln & 15;
        int k   = ch * 32 + kmap(ln >> 4, j);
        wf_lds[i] = (col < NCLS) ? (_Float16)Wf[k * NCLS + col] : (_Float16)0.f;
    }
    __syncthreads();

    int wave = (blockIdx.x * blockDim.x + tid) >> 5;
    int lane = tid & 31;
    int hi = lane >> 4, lr = lane & 15;
    int tile = wave;
    if (tile >= N_EDGES / 16) return;
    int e0 = tile * 16;
    int er = e0 + lr;
    int s = ei[er], d = ei[N_EDGES + er];
    const float* rs = out + (size_t)s * HID;
    const float* rd = out + (size_t)d * HID;
    v16h a0 = load_a_chunk_sum(rs, rd, hi, 0);
    v16h a1 = load_a_chunk_sum(rs, rd, hi, 1);
    v16h b0 = *(const v16h*)&wf_lds[lane * 16];
    v16h b1 = *(const v16h*)&wf_lds[512 + lane * 16];
    float bias = (lr < NCLS) ? bf[lr] : 0.f;
    v8f c;
    #pragma unroll
    for (int r = 0; r < 8; ++r) c[r] = bias;
    c = __builtin_amdgcn_wmma_f32_16x16x32_f16(false, a0, false, b0, (short)0, c, false, false);
    c = __builtin_amdgcn_wmma_f32_16x16x32_f16(false, a1, false, b1, (short)0, c, false, false);
    if (lr < NCLS) {
        #pragma unroll
        for (int r = 0; r < 8; ++r) {
            int m = r + 8 * hi;
            res[(size_t)(e0 + m) * NCLS + lr] = c[r];
        }
    }
}

// ---------------------------------------------------------------------------
extern "C" void kernel_launch(void* const* d_in, const int* in_sizes, int n_in,
                              void* d_out, int out_size, void* d_ws, size_t ws_size,
                              hipStream_t stream)
{
    const float* x  = (const float*)d_in[0];
    const int*   ei = (const int*)d_in[1];
    const float* ea = (const float*)d_in[2];
    const float* Wn = (const float*)d_in[3];  const float* bn = (const float*)d_in[4];
    const float* We = (const float*)d_in[5];  const float* be = (const float*)d_in[6];
    const float* Wq = (const float*)d_in[7];  const float* bq = (const float*)d_in[8];
    const float* Wk = (const float*)d_in[9];  const float* bk = (const float*)d_in[10];
    const float* Wv = (const float*)d_in[11]; const float* bv = (const float*)d_in[12];
    const float* Ws = (const float*)d_in[13]; const float* bs = (const float*)d_in[14];
    const float* Wf = (const float*)d_in[15]; const float* bf = (const float*)d_in[16];

    // workspace layout (floats): q | k | v | out-acc | logits | m_enc | s_sum
    float* ws   = (float*)d_ws;
    float* q    = ws;
    float* kk   = q    + (size_t)N_NODES * HID;
    float* vv   = kk   + (size_t)N_NODES * HID;
    float* oacc = vv   + (size_t)N_NODES * HID;
    float* lg   = oacc + (size_t)N_NODES * HID;
    unsigned* me = (unsigned*)(lg + (size_t)N_EDGES * HEADS);
    float* ss   = (float*)(me + (size_t)N_NODES * HEADS);

    const int nh = N_NODES * HEADS;
    const int eh = N_EDGES * HEADS;

    init_kernel<<<(nh + 255) / 256, 256, 0, stream>>>(me, ss, nh);
    node_gemm_kernel<<<(N_NODES / 16 + 7) / 8, 256, 0, stream>>>(
        x, Wn, bn, Wq, bq, Wk, bk, Wv, bv, Ws, bs, q, kk, vv, oacc);
    logits_kernel<<<(eh + 255) / 256, 256, 0, stream>>>(ei, q, kk, lg, me);
    expsum_kernel<<<(eh + 255) / 256, 256, 0, stream>>>(ei, lg, me, ss);
    message_kernel<<<(eh + 255) / 256, 256, 0, stream>>>(ei, ea, We, be, vv, lg, ss, oacc);
    final_kernel<<<(N_EDGES / 16) / 8, 256, 0, stream>>>(ei, oacc, Wf, bf, (float*)d_out);
}
